// DictBlock_11347303596597
// MI455X (gfx1250) — matrix operand order repaired and checked by reading, not compiled, not run on hardware
//
#include <hip/hip_runtime.h>
#include <math.h>

// MI455X (gfx1250) FISTA conv sparse coding.
// Convs are implicit GEMM on v_wmma_f32_16x16x32_bf16 (wave32, 16x16 tiles,
// K chunks of 32 = pairs of (tap, 16 input channels)). All elementwise work
// fused into conv epilogues; losses via deterministic tree reductions.

typedef __attribute__((ext_vector_type(16))) __bf16 v16bf;
typedef __attribute__((ext_vector_type(8)))  float  v8f;

namespace {
constexpr int   N_   = 32;
constexpr int   H_   = 112;
constexpr int   W_   = 112;
constexpr float STEP_     = 0.1f;
constexpr float PROX_S    = 1.0f / 1.01f;   // 1/(1+mu*step)
constexpr float PROX_THR  = 0.01f / 1.01f;  // lmbd*step*s
constexpr int   RED_BLOCKS = 1024;
}

// ---------------- weight normalize + A-fragment packing ----------------
// A packing used here (16x32, contiguous K; B loads match): lane<16 holds row M=lane, K in {0..15};
// lane>=16 holds K in {16..31}. Element e -> Klocal:
//   Klocal = ((lane>=16)?16:0) | e
__global__ __launch_bounds__(256) void pack_weights_kernel(
    const float* __restrict__ w, __bf16* __restrict__ packF, __bf16* __restrict__ packT) {
  __shared__ float scale[64];
  const int tid = threadIdx.x;
  if (tid < 64) {
    float s = 0.f;
    for (int k = 0; k < 144; ++k) { float v = w[tid * 144 + k]; s += v * v; }
    scale[tid] = 1.0f / fmaxf(sqrtf(s), 1e-12f);
  }
  __syncthreads();
  // forward conv: [cot(4)][ch(5)][lane(32)][e(16)], chunk halves = taps 2ch, 2ch+1
  for (int i = tid; i < 4 * 5 * 32 * 16; i += 256) {
    int e = i & 15, lane = (i >> 4) & 31, ch = (i >> 9) % 5, cot = i / (5 * 512);
    int kl = ((lane >> 4) << 4) | e;
    int half = kl >> 4, kk = kl & 15;
    int tap = 2 * ch + half;
    int co  = cot * 16 + (lane & 15);
    float v = 0.f;
    if (tap <= 8) v = w[(co * 16 + kk) * 9 + tap] * scale[co];  // zero-pad tap 9
    packF[i] = (__bf16)v;
  }
  // conv_t (flipped, transposed): [ch(18)][lane(32)][e(16)], chunk = (tap=ch/2, ci half)
  for (int i = tid; i < 18 * 32 * 16; i += 256) {
    int e = i & 15, lane = (i >> 4) & 31, ch = i >> 9;
    int kl = ((lane >> 4) << 4) | e;
    int half = kl >> 4, kk = kl & 15;
    int tap = ch >> 1;
    int co  = lane & 15;                         // conv_t out ch = w input ch
    int ci  = (ch & 1) * 32 + half * 16 + kk;    // conv_t in ch = w output ch
    float v = w[(ci * 16 + co) * 9 + (8 - tap)] * scale[ci];
    packT[i] = (__bf16)v;
  }
}

// ---------------- implicit-GEMM conv with fused FISTA epilogue ----------------
// EPI=0: out = relu(prox(alpha*aux0 + beta*aux1 + step*acc))   (forward conv)
// EPI=1: out = aux0 - acc                                      (residual conv_t)
template <int CIN, int COUT, int SEGW, int NCH, int EPI>
__global__ __launch_bounds__(256) void conv_fista_kernel(
    const float* __restrict__ in, const __bf16* __restrict__ packW,
    const float* __restrict__ aux0, const float* __restrict__ aux1,
    float* __restrict__ out, float alpha, float beta) {
  constexpr int NCOT = COUT / 16;
  constexpr int NPG  = SEGW / 16;
  constexpr int SEGS = (W_ + SEGW - 1) / SEGW;
  constexpr int C    = SEGW + 2;
  __shared__ __align__(32) __bf16 tile[3 * C * CIN];  // [row][col][ci], ci contiguous

  const int bid = blockIdx.x;
  const int seg = bid % SEGS;
  const int h   = (bid / SEGS) % H_;
  const int n   = bid / (SEGS * H_);
  const int w0  = seg * SEGW;
  const int tid = threadIdx.x;

  // Stage fp32 -> bf16 halo tile into LDS (coalesced along cols, zero-padded).
  for (int i = tid; i < 3 * C * CIN; i += 256) {
    int col = i % C; int t = i / C; int ci = t % CIN; int r = t / CIN;
    int gh = h - 1 + r, gw = w0 - 1 + col;
    float v = 0.f;
    if (gh >= 0 && gh < H_ && gw >= 0 && gw < W_)
      v = in[((n * CIN + ci) * H_ + gh) * W_ + gw];
    tile[(r * C + col) * CIN + ci] = (__bf16)v;
  }
  __syncthreads();

  const int wv   = tid >> 5;
  const int lane = tid & 31;
  if (wv >= NCOT * NPG) return;   // wave-uniform; EXEC stays all-ones for WMMA
  const int cot = wv / NPG;
  const int pg  = wv % NPG;
  const int p   = lane & 15;
  const int hi  = lane >> 4;      // B half: lanes<16 carry K 0..15, lanes>=16 K 16..31

  v8f acc = {};
  for (int ch = 0; ch < NCH; ++ch) {
    int tap, cb;
    if constexpr (CIN == 16) { int tr = 2 * ch + hi; tap = tr > 8 ? 8 : tr; cb = 0; }
    else                     { tap = ch >> 1; cb = (ch & 1) * 32 + hi * 16; }
    v16bf a = *(const v16bf*)(packW + ((cot * NCH + ch) * 32 + lane) * 16);
    int kh = tap / 3, kw = tap % 3;
    v16bf b = *(const v16bf*)&tile[(kh * C + (pg * 16 + p + kw)) * CIN + cb];
    acc = __builtin_amdgcn_wmma_f32_16x16x32_bf16(false, a, false, b,
                                                  (short)0, acc, false, false);
  }

  const int wpix = w0 + pg * 16 + p;
  if (wpix >= W_) return;
  // D layout: VGPR j -> M=j (lanes<16) / M=j+8 (lanes>=16), N = lane%16
  const int co0 = cot * 16 + hi * 8;
#pragma unroll
  for (int j = 0; j < 8; ++j) {
    int idx = ((n * COUT + (co0 + j)) * H_ + h) * W_ + wpix;
    float av = acc[j];
    if constexpr (EPI == 0) {
      float v = STEP_ * av;
      if (alpha != 0.f) v += alpha * aux0[idx];
      if (beta  != 0.f) v += beta  * aux1[idx];
      v = PROX_S * v - PROX_THR;             // relu(softshrink) collapses to this
      out[idx] = v > 0.f ? v : 0.f;
    } else {
      out[idx] = aux0[idx] - av;             // r = x - conv_t(c)
    }
  }
}

// ---------------- deterministic loss reductions ----------------
__global__ __launch_bounds__(256) void reduce_partial_kernel(
    const float* __restrict__ v, int n, int mode, float* __restrict__ partial) {
  __shared__ float sm[256];
  float s = 0.f;
  for (int i = blockIdx.x * 256 + threadIdx.x; i < n; i += gridDim.x * 256) {
    float x = v[i];
    s += mode ? (0.1f * fabsf(x) + 0.05f * x * x) : x * x;
  }
  sm[threadIdx.x] = s;
  __syncthreads();
  for (int st = 128; st > 0; st >>= 1) {
    if (threadIdx.x < st) sm[threadIdx.x] += sm[threadIdx.x + st];
    __syncthreads();
  }
  if (threadIdx.x == 0) partial[blockIdx.x] = sm[0];
}

__global__ __launch_bounds__(256) void final_reduce_kernel(
    const float* __restrict__ pR, const float* __restrict__ pC, float* __restrict__ loss) {
  __shared__ float sm[256];
  float s = 0.f;
  for (int i = threadIdx.x; i < RED_BLOCKS; i += 256) s += pR[i];
  sm[threadIdx.x] = s; __syncthreads();
  for (int st = 128; st > 0; st >>= 1) {
    if (threadIdx.x < st) sm[threadIdx.x] += sm[threadIdx.x + st];
    __syncthreads();
  }
  if (threadIdx.x == 0) loss[0] = sm[0];
  __syncthreads();
  s = 0.f;
  for (int i = threadIdx.x; i < RED_BLOCKS; i += 256) s += pC[i];
  sm[threadIdx.x] = s; __syncthreads();
  for (int st = 128; st > 0; st >>= 1) {
    if (threadIdx.x < st) sm[threadIdx.x] += sm[threadIdx.x + st];
    __syncthreads();
  }
  if (threadIdx.x == 0) loss[1] = sm[0];
}

// ---------------- host orchestration ----------------
extern "C" void kernel_launch(void* const* d_in, const int* in_sizes, int n_in,
                              void* d_out, int out_size, void* d_ws, size_t ws_size,
                              hipStream_t stream) {
  (void)in_sizes; (void)n_in; (void)out_size; (void)ws_size;
  const float* x = (const float*)d_in[0];
  const float* w = (const float*)d_in[1];
  float* outF = (float*)d_out;
  const size_t Nc = (size_t)N_ * 64 * H_ * W_;   // 25,690,112
  const size_t Nr = (size_t)N_ * 16 * H_ * W_;   //  6,422,528

  char* ws = (char*)d_ws;
  size_t off = 0;
  auto carve = [&](size_t bytes) -> char* {
    char* pp = ws + off;
    off = (off + bytes + 255) & ~(size_t)255;
    return pp;
  };
  float*  c0    = (float*)carve(Nc * 4);
  float*  rbuf  = (float*)carve(Nr * 4);
  __bf16* packF = (__bf16*)carve((size_t)4 * 5 * 32 * 16 * 2);
  __bf16* packT = (__bf16*)carve((size_t)18 * 32 * 16 * 2);
  float*  pR    = (float*)carve((size_t)RED_BLOCKS * 4);
  float*  pC    = (float*)carve((size_t)RED_BLOCKS * 4);

  float* cbuf[2] = { c0, outF };  // iter i writes cbuf[i&1]; iter 9 lands in d_out

  pack_weights_kernel<<<1, 256, 0, stream>>>(w, packF, packT);

  const int gridF = N_ * H_ * 4;  // SEGW=32  -> 4 segments/row
  const int gridT = N_ * H_;      // SEGW=112 -> 1 segment/row

  // i == 0: c = relu(prox(step * conv(x, w)))
  conv_fista_kernel<16, 64, 32, 5, 0><<<gridF, 256, 0, stream>>>(
      x, packF, c0, c0, cbuf[0], 0.f, 0.f);

  double t = (sqrt(5.0) + 1.0) / 2.0;
  for (int i = 1; i < 10; ++i) {
    float alpha, beta;
    if (i == 1) { alpha = 1.f; beta = 0.f; }
    else {
      double tp = t;
      t = (sqrt(1.0 + 4.0 * tp * tp) + 1.0) / 2.0;
      alpha = (float)((tp + t - 1.0) / t);
      beta  = (float)((1.0 - tp) / t);
    }
    float* ccur = cbuf[(i + 1) & 1];
    float* cnew = cbuf[i & 1];  // on entry (i>=2) holds c_prev; epilogue read-then-write per element
    // r = x - conv_t(c_cur, w)
    conv_fista_kernel<64, 16, 112, 18, 1><<<gridT, 256, 0, stream>>>(
        ccur, packT, x, x, rbuf, 0.f, 0.f);
    // c = relu(prox(alpha*c_cur + beta*c_prev + step*conv(r, w)))
    conv_fista_kernel<16, 64, 32, 5, 0><<<gridF, 256, 0, stream>>>(
        rbuf, packF, ccur, cnew, cnew, alpha, beta);
  }

  // final residual: r = x - conv_t(c, w) with c in d_out
  conv_fista_kernel<64, 16, 112, 18, 1><<<gridT, 256, 0, stream>>>(
      outF, packT, x, x, rbuf, 0.f, 0.f);

  reduce_partial_kernel<<<RED_BLOCKS, 256, 0, stream>>>(rbuf, (int)Nr, 0, pR);
  reduce_partial_kernel<<<RED_BLOCKS, 256, 0, stream>>>(outF, (int)Nc, 1, pC);
  final_reduce_kernel<<<1, 256, 0, stream>>>(pR, pC, outF + Nc);
}
